// Answer_filtering_module_26783416058216
// MI455X (gfx1250) — matrix operand order repaired and checked by reading, not compiled, not run on hardware
//
#include <hip/hip_runtime.h>
#include <math.h>

#define B_    128
#define T_    32
#define NENT  200000
#define EDIM  400

typedef __attribute__((ext_vector_type(16))) __bf16    v16bf;
typedef __attribute__((ext_vector_type(16))) short     v16s;
typedef __attribute__((ext_vector_type(8)))  unsigned  v8u;
typedef __attribute__((ext_vector_type(8)))  float     v8f;

// ---- fp32 workspace layout (float offsets) ----
// acc: [0]=bce [1]=cl [2..3]=head mean [4..5]=head var [6..7]=score mean [8..9]=score var
#define OFF_ACC    0L
#define OFF_E      64L        // 4096
#define OFF_C      4160L      // 2 dirs * 128*256 = 65536
#define OFF_HBF    69696L     // bf16 h ping-pong: 4*32768 shorts = 65536 floats
#define ZERO_N     135232
#define OFF_HCAT   135232L    // 128*32*512
#define OFF_HBSCAN 2232384L   // 128*32*256
#define OFF_REL    3280960L   // 128*400
#define OFF_RES    3332160L   // 128*200
#define OFF_IMS    3357760L   // 128*200
#define OFF_ANS    3383360L   // 128*400
#define FLOAT_TOTAL 3434560L

// ---- bf16 workspace layout (short offsets, base = ws + FLOAT_TOTAL) ----
#define SOFF_WHH_F 0L         // 1024*256
#define SOFF_WIH_F 262144L
#define SOFF_WHH_B 524288L
#define SOFF_WIH_B 786432L
#define SOFF_WA    1048576L   // transposed: 1024 rows n x 512 cols k
#define SOFF_WFC1  1572864L   // 512*512
#define SOFF_WFC2  1835008L   // 400*512
#define SOFF_HCATB 2039808L   // 128*32*512
#define SOFF_CTXB  4136960L   // 128*512
#define SOFF_HIDB  4202496L   // 128*512
#define SOFF_SNB   4268032L   // 128*416

__device__ inline short f2bfs(float f) {
    __bf16 b = (__bf16)f;
    return __builtin_bit_cast(short, b);
}

// CDNA5 packed convert: 2 x fp32 -> packed 2 x bf16 in one VALU op
__device__ inline unsigned cvt_pk_bf16(float lo, float hi) {
    unsigned d;
    asm("v_cvt_pk_bf16_f32 %0, %1, %2" : "=v"(d) : "v"(lo), "v"(hi));
    return d;
}

__device__ inline v8f vzero() {
    v8f z;
#pragma unroll
    for (int i = 0; i < 8; ++i) z[i] = 0.f;
    return z;
}

// A fragment (16x32 bf16 MxK) from a contiguous bf16 row:
// lane holds m=lane%16; k runs [base..base+15], base=16*(lane>=16)
__device__ inline v16bf loadA_bf(const short* __restrict__ row, int kb, int lane) {
    int base = kb + ((lane >= 16) ? 16 : 0);
    v16s s;
#pragma unroll
    for (int i = 0; i < 8; ++i) s[i] = row[base + i];
#pragma unroll
    for (int i = 0; i < 8; ++i) s[8 + i] = row[base + 8 + i];
    return __builtin_bit_cast(v16bf, s);
}

// A fragment converted on the fly from an fp32 row (packed cvt, 8 VALU ops)
__device__ inline v16bf loadA_f32(const float* __restrict__ row, int kb, int lane) {
    int base = kb + ((lane >= 16) ? 16 : 0);
    v8u u;
#pragma unroll
    for (int i = 0; i < 4; ++i)
        u[i] = cvt_pk_bf16(row[base + 2 * i], row[base + 2 * i + 1]);
#pragma unroll
    for (int i = 0; i < 4; ++i)
        u[4 + i] = cvt_pk_bf16(row[base + 8 + 2 * i], row[base + 8 + 2 * i + 1]);
    return __builtin_bit_cast(v16bf, u);
}

// B fragment (32x16 bf16 KxN) for NT gemm from contiguous bf16 row B[n][*]:
// lane holds n=lane%16, k = k0+i, k0 = kb + 16*(lane>=16) -> one 32B load
__device__ inline v16bf loadB_bf(const short* __restrict__ row, int kb, int lane) {
    int k0 = kb + ((lane >= 16) ? 16 : 0);
    v16s s;
#pragma unroll
    for (int i = 0; i < 16; ++i) s[i] = row[k0 + i];
    return __builtin_bit_cast(v16bf, s);
}

__device__ inline v8f wmma_bf16(v16bf a, v16bf b, v8f c) {
    return __builtin_amdgcn_wmma_f32_16x16x32_bf16(false, a, false, b, (short)0, c,
                                                   false, false);
}

__device__ inline float waveSum(float v) {
    for (int m = 16; m >= 1; m >>= 1) v += __shfl_xor(v, m, 32);
    return v;
}

// ---------------- kernels ----------------

__global__ void k_zero(float* p, int n) {
    int i = blockIdx.x * 256 + threadIdx.x;
    if (i < n) p[i] = 0.f;
}

__global__ void k_f2bf(const float* __restrict__ src, short* __restrict__ dst, int n) {
    int i = blockIdx.x * 256 + threadIdx.x;
    if (i < n) dst[i] = f2bfs(src[i]);
}

// fp32 [K x N] row-major -> bf16 [N x K] (NT layout) with conversion
__global__ void k_f2bf_t(const float* __restrict__ src, short* __restrict__ dst,
                         int K, int N) {
    int i = blockIdx.x * 256 + threadIdx.x;
    if (i >= K * N) return;
    int n = i / K, k = i % K;
    dst[i] = f2bfs(src[(long)k * N + n]);
}

// One LSTM time step, both directions. g = [h|x] @ [Whh|Wih]^T + b, gates fused.
__global__ __launch_bounds__(32)
void k_lstm_step(const float* __restrict__ wordE, const int* __restrict__ question,
                 const int* __restrict__ qlen,
                 const float* __restrict__ bih_f, const float* __restrict__ bhh_f,
                 const float* __restrict__ bih_b, const float* __restrict__ bhh_b,
                 float* __restrict__ ws, int t)
{
    int lane = threadIdx.x;
    int jt = blockIdx.x;   // hidden tile 0..15
    int mt = blockIdx.y;   // batch tile 0..7
    int dir = blockIdx.z;  // 0 fwd, 1 bwd
    short* sws = (short*)(ws + FLOAT_TOTAL);
    const short* Whh = sws + (dir ? SOFF_WHH_B : SOFF_WHH_F);
    const short* Wih = sws + (dir ? SOFF_WIH_B : SOFF_WIH_F);
    const float* bih = dir ? bih_b : bih_f;
    const float* bhh = dir ? bhh_b : bhh_f;

    short* hb = (short*)(ws + OFF_HBF) + (long)dir * 65536;
    const short* hprev = hb + (t & 1) * 32768;
    short* hnext = hb + ((t + 1) & 1) * 32768;
    float* cbuf = ws + OFF_C + (long)dir * 32768;

    int mA = mt * 16 + (lane & 15);
    int len = qlen[mA];
    int teff = dir ? ((t < len) ? (len - 1 - t) : t) : t;
    const float* xrow = wordE + (long)question[mA * T_ + teff] * 256;
    const short* hrow = hprev + (long)mA * 256;

    v8f a0 = vzero(), a1 = vzero(), a2 = vzero(), a3 = vzero();
    int jbase = jt * 16 + (lane & 15);
    for (int kc = 0; kc < 8; ++kc) {            // h @ Whh^T
        int kb = kc * 32;
        v16bf a = loadA_bf(hrow, kb, lane);
        a0 = wmma_bf16(a, loadB_bf(Whh + (long)(0 * 256 + jbase) * 256, kb, lane), a0);
        a1 = wmma_bf16(a, loadB_bf(Whh + (long)(1 * 256 + jbase) * 256, kb, lane), a1);
        a2 = wmma_bf16(a, loadB_bf(Whh + (long)(2 * 256 + jbase) * 256, kb, lane), a2);
        a3 = wmma_bf16(a, loadB_bf(Whh + (long)(3 * 256 + jbase) * 256, kb, lane), a3);
    }
    for (int kc = 0; kc < 8; ++kc) {            // x @ Wih^T (on-the-fly cvt of x)
        int kb = kc * 32;
        v16bf a = loadA_f32(xrow, kb, lane);
        a0 = wmma_bf16(a, loadB_bf(Wih + (long)(0 * 256 + jbase) * 256, kb, lane), a0);
        a1 = wmma_bf16(a, loadB_bf(Wih + (long)(1 * 256 + jbase) * 256, kb, lane), a1);
        a2 = wmma_bf16(a, loadB_bf(Wih + (long)(2 * 256 + jbase) * 256, kb, lane), a2);
        a3 = wmma_bf16(a, loadB_bf(Wih + (long)(3 * 256 + jbase) * 256, kb, lane), a3);
    }
    int j = jbase;
    float bi0 = bih[j] + bhh[j];
    float bf0 = bih[256 + j] + bhh[256 + j];
    float bg0 = bih[512 + j] + bhh[512 + j];
    float bo0 = bih[768 + j] + bhh[768 + j];
#pragma unroll
    for (int v = 0; v < 8; ++v) {
        int m = mt * 16 + v + 8 * (lane >> 4);
        float si = 1.f / (1.f + expf(-(a0[v] + bi0)));
        float sf = 1.f / (1.f + expf(-(a1[v] + bf0)));
        float gg = a2[v] + bg0;
        float so = 1.f / (1.f + expf(-(a3[v] + bo0)));
        long ci = (long)m * 256 + j;
        float c = cbuf[ci];
        c = sf * c + si * tanhf(gg);
        cbuf[ci] = c;
        float h = so * tanhf(c);
        hnext[ci] = f2bfs(h);
        if (dir == 0) ws[OFF_HCAT + ((long)m * T_ + t) * 512 + j] = h;
        else          ws[OFF_HBSCAN + ((long)m * T_ + t) * 256 + j] = h;
    }
}

// un-reverse backward states into h_concat, apply mask, emit fp32 + bf16 copies
__global__ __launch_bounds__(256)
void k_unrev(const int* __restrict__ qlen, float* __restrict__ ws)
{
    int r = blockIdx.x;          // b*T + t
    int b = r >> 5, t = r & 31;
    int len = qlen[b];
    bool msk = t < len;
    int rv = msk ? (len - 1 - t) : t;
    short* hcatb = (short*)(ws + FLOAT_TOTAL) + SOFF_HCATB;
    for (int j = threadIdx.x; j < 512; j += 256) {
        long o = (long)r * 512 + j;
        float v;
        if (j < 256) {
            v = msk ? ws[OFF_HCAT + o] : 0.f;
        } else {
            v = msk ? ws[OFF_HBSCAN + ((long)b * T_ + rv) * 256 + (j - 256)] : 0.f;
        }
        ws[OFF_HCAT + o] = v;
        hcatb[o] = f2bfs(v);
    }
}

// e[b,t] = tanh(h @ Wa) . va   (NT gemm on transposed Wa, fused va-dot epilogue)
__global__ __launch_bounds__(32)
void k_attn(const float* __restrict__ va, float* __restrict__ ws)
{
    int lane = threadIdx.x;
    int nt = blockIdx.x;   // 0..63
    int mt = blockIdx.y;   // 0..255
    short* sws = (short*)(ws + FLOAT_TOTAL);
    const short* rowA = sws + SOFF_HCATB + (long)(mt * 16 + (lane & 15)) * 512;
    int n = nt * 16 + (lane & 15);
    const short* rowB = sws + SOFF_WA + (long)n * 512;
    v8f acc = vzero();
    for (int kb = 0; kb < 512; kb += 32)
        acc = wmma_bf16(loadA_bf(rowA, kb, lane), loadB_bf(rowB, kb, lane), acc);
    float* e = ws + OFF_E;
#pragma unroll
    for (int v = 0; v < 8; ++v) {
        float val = tanhf(acc[v]) * va[n];
        for (int mk = 8; mk >= 1; mk >>= 1) val += __shfl_xor(val, mk, 32);
        if ((lane & 15) == 0) {
            int m = mt * 16 + v + 8 * (lane >> 4);
            atomicAdd(e + m, val);
        }
    }
}

__global__ __launch_bounds__(256)
void k_softmax_ctx(const int* __restrict__ qlen, float* __restrict__ ws)
{
    int b = blockIdx.x;
    __shared__ float al[T_];
    int tid = threadIdx.x;
    if (tid < 32) {
        int len = qlen[b];
        float v = (tid < len) ? ws[OFF_E + b * T_ + tid] : -1e9f;
        float mx = v;
        for (int m = 16; m >= 1; m >>= 1) mx = fmaxf(mx, __shfl_xor(mx, m, 32));
        float ex = expf(v - mx);
        float s = ex;
        for (int m = 16; m >= 1; m >>= 1) s += __shfl_xor(s, m, 32);
        al[tid] = ex / s;
    }
    __syncthreads();
    short* ctxb = (short*)(ws + FLOAT_TOTAL) + SOFF_CTXB;
    for (int d = tid; d < 512; d += 256) {
        float s = 0.f;
        for (int t2 = 0; t2 < T_; ++t2)
            s += al[t2] * ws[OFF_HCAT + ((long)b * T_ + t2) * 512 + d];
        ctxb[(long)b * 512 + d] = f2bfs(s);
    }
}

// generic bf16 NT gemm: fp32 C and/or bf16 C outputs, optional bias+relu
__global__ __launch_bounds__(32)
void k_gemm_nt(const short* __restrict__ A, int lda,
               const short* __restrict__ Bm, int ldb,
               const float* __restrict__ bias,
               float* __restrict__ C, short* __restrict__ Cbf, int ldc,
               int K, int relu)
{
    int lane = threadIdx.x;
    int nt = blockIdx.x, mt = blockIdx.y;
    const short* rowA = A + (long)(mt * 16 + (lane & 15)) * lda;
    const short* rowB = Bm + (long)(nt * 16 + (lane & 15)) * ldb;
    v8f acc = vzero();
    for (int kb = 0; kb < K; kb += 32)
        acc = wmma_bf16(loadA_bf(rowA, kb, lane), loadB_bf(rowB, kb, lane), acc);
#pragma unroll
    for (int v = 0; v < 8; ++v) {
        int m = mt * 16 + v + 8 * (lane >> 4);
        int n = nt * 16 + (lane & 15);
        float x = acc[v];
        if (bias) x += bias[n];
        if (relu) x = fmaxf(x, 0.f);
        if (C)   C[(long)m * ldc + n] = x;
        if (Cbf) Cbf[(long)m * ldc + n] = f2bfs(x);
    }
}

// mean/var per 2 channels: which=0 -> gathered head rows, which=1 -> re_s/im_s
__global__ __launch_bounds__(256)
void k_stats(const float* __restrict__ E, const int* __restrict__ head,
             float* __restrict__ ws, int which)
{
    __shared__ float s1[256], s2[256];
    for (int ch = 0; ch < 2; ++ch) {
        float s = 0.f, ss = 0.f;
        for (int i = threadIdx.x; i < 25600; i += 256) {
            float v;
            if (which) v = ws[(ch ? OFF_IMS : OFF_RES) + i];
            else {
                int b = i / 200, j = i % 200;
                v = E[(long)head[b] * EDIM + ch * 200 + j];
            }
            s += v; ss += v * v;
        }
        s1[threadIdx.x] = s; s2[threadIdx.x] = ss;
        __syncthreads();
        for (int st = 128; st >= 1; st >>= 1) {
            if (threadIdx.x < st) {
                s1[threadIdx.x] += s1[threadIdx.x + st];
                s2[threadIdx.x] += s2[threadIdx.x + st];
            }
            __syncthreads();
        }
        if (threadIdx.x == 0) {
            float mean = s1[0] / 25600.f;
            ws[2 + which * 4 + ch] = mean;
            ws[4 + which * 4 + ch] = s2[0] / 25600.f - mean * mean;
        }
        __syncthreads();
    }
}

__global__ __launch_bounds__(256)
void k_res_ims(const float* __restrict__ E, const int* __restrict__ head,
               const float* __restrict__ hw, const float* __restrict__ hb,
               float* __restrict__ ws)
{
    int i = blockIdx.x * 256 + threadIdx.x;
    if (i >= 25600) return;
    int b = i / 200, j = i % 200;
    long e0 = (long)head[b] * EDIM;
    float r0 = rsqrtf(ws[4] + 1e-5f);
    float r1 = rsqrtf(ws[5] + 1e-5f);
    float reh = (E[e0 + j] - ws[2]) * r0 * hw[0] + hb[0];
    float imh = (E[e0 + 200 + j] - ws[3]) * r1 * hw[1] + hb[1];
    float rer = ws[OFF_REL + (long)b * 400 + j];
    float imr = ws[OFF_REL + (long)b * 400 + 200 + j];
    ws[OFF_RES + i] = reh * rer - imh * imr;
    ws[OFF_IMS + i] = reh * imr + imh * rer;
}

// score-BN normalize and pack [sn0|sn1|zero-pad] bf16 rows of width 416
__global__ __launch_bounds__(256)
void k_make_sn(const float* __restrict__ sw, const float* __restrict__ sb,
               float* __restrict__ ws)
{
    int i = blockIdx.x * 256 + threadIdx.x;
    if (i >= B_ * 416) return;
    int b = i / 416, col = i % 416;
    float out = 0.f;
    if (col < 400) {
        int ch = col / 200, j = col % 200;
        float v = ch == 0 ? ws[OFF_RES + (long)b * 200 + j]
                          : ws[OFF_IMS + (long)b * 200 + j];
        out = (v - ws[6 + ch]) * rsqrtf(ws[8 + ch] + 1e-5f) * sw[ch] + sb[ch];
    }
    short* snb = (short*)(ws + FLOAT_TOTAL) + SOFF_SNB;
    snb[i] = f2bfs(out);
}

// fused score GEMM (128x200000, K=400) + sigmoid/BCE (1 exp + 1 log1p per elem)
__global__ __launch_bounds__(256)
void k_score_bce(const float* __restrict__ E, const float* __restrict__ tail,
                 float* __restrict__ ws)
{
    int tid = threadIdx.x;
    int lane = tid & 31;
    int w = tid >> 5;       // m-tile 0..7
    int nt = blockIdx.x;    // entity tile
    const short* snb = (const short*)(ws + FLOAT_TOTAL) + SOFF_SNB;
    const short* rowA = snb + (long)(w * 16 + (lane & 15)) * 416;
    const float* rowB = E + (long)(nt * 16 + (lane & 15)) * EDIM;
    v8f acc = vzero();
    for (int kc = 0; kc < 13; ++kc) {
        int kb = kc * 32;
        v16bf a = loadA_bf(rowA, kb, lane);
        int k0 = kb + ((lane >= 16) ? 16 : 0);
        v8u u;
#pragma unroll
        for (int i = 0; i < 8; ++i) {
            int k = k0 + 2 * i;
            float lo = (k < EDIM) ? rowB[k] : 0.f;
            float hi = (k + 1 < EDIM) ? rowB[k + 1] : 0.f;
            u[i] = cvt_pk_bf16(lo, hi);
        }
        v16bf b = __builtin_bit_cast(v16bf, u);
        acc = wmma_bf16(a, b, acc);
    }
    float local = 0.f;
    int ng = nt * 16 + (lane & 15);
#pragma unroll
    for (int v = 0; v < 8; ++v) {
        int m = w * 16 + v + 8 * (lane >> 4);
        float x = acc[v];
        float tv = tail[(long)m * NENT + ng];
        // log(sigmoid(x)) = -log1p(e^-x); log(1-sigmoid(x)) = -x - log1p(e^-x)
        float l = log1pf(expf(-x));
        float lp = fmaxf(-l, -100.f);
        float l1 = fmaxf(-x - l, -100.f);
        local -= tv * lp + (1.f - tv) * l1;
    }
    local = waveSum(local);
    if (lane == 0) atomicAdd(ws + 0, local);
}

__global__ __launch_bounds__(256)
void k_argmax_ans(const float* __restrict__ tail, const float* __restrict__ E,
                  float* __restrict__ ws)
{
    int b = blockIdx.x;
    __shared__ float bv[256];
    __shared__ int bi[256];
    float best = -1e30f;
    int besti = 0;
    for (int i = threadIdx.x; i < NENT; i += 256) {
        float v = tail[(long)b * NENT + i];
        if (v > best) { best = v; besti = i; }
    }
    bv[threadIdx.x] = best; bi[threadIdx.x] = besti;
    __syncthreads();
    for (int st = 128; st >= 1; st >>= 1) {
        if (threadIdx.x < st) {
            float ov = bv[threadIdx.x + st]; int oi = bi[threadIdx.x + st];
            if (ov > bv[threadIdx.x] || (ov == bv[threadIdx.x] && oi < bi[threadIdx.x])) {
                bv[threadIdx.x] = ov; bi[threadIdx.x] = oi;
            }
        }
        __syncthreads();
    }
    long idx = bi[0];
    for (int k = threadIdx.x; k < EDIM; k += 256)
        ws[OFF_ANS + (long)b * EDIM + k] = E[idx * EDIM + k];
}

__global__ __launch_bounds__(256)
void k_contrastive(const float* __restrict__ E, const int* __restrict__ pos,
                   const int* __restrict__ neg, float* __restrict__ ws)
{
    int b = blockIdx.x;
    __shared__ float d[220];
    __shared__ float red[256];
    int tid = threadIdx.x;
    if (tid < 220) {
        int row = (tid < 20) ? pos[b * 20 + tid] : neg[b * 200 + (tid - 20)];
        const float* er = E + (long)row * EDIM;
        const float* ar = ws + OFF_ANS + (long)b * EDIM;
        float s = 0.f;
        for (int k = 0; k < EDIM; ++k) {
            float df = ar[k] - er[k] + 1e-6f;
            s += df * df;
        }
        d[tid] = sqrtf(s);
    }
    __syncthreads();
    float local = 0.f;
    for (int i = tid; i < 4000; i += 256) {
        int p = i / 200, q = i % 200;
        float v = d[p] + 1e-4f - d[20 + q];
        if (v > 0.f) local += v;
    }
    red[tid] = local;
    __syncthreads();
    for (int st = 128; st >= 1; st >>= 1) {
        if (tid < st) red[tid] += red[tid + st];
        __syncthreads();
    }
    if (tid == 0) atomicAdd(ws + 1, red[0]);
}

__global__ void k_finalize(const float* __restrict__ ws, float* __restrict__ out)
{
    if (threadIdx.x == 0 && blockIdx.x == 0)
        out[0] = ws[0] + 5e-4f * ws[1];
}

// ---------------- host launcher ----------------
extern "C" void kernel_launch(void* const* d_in, const int* in_sizes, int n_in,
                              void* d_out, int out_size, void* d_ws, size_t ws_size,
                              hipStream_t stream)
{
    (void)in_sizes; (void)n_in; (void)out_size; (void)ws_size;
    const int*   question  = (const int*)d_in[0];
    const int*   qlen      = (const int*)d_in[1];
    const int*   head      = (const int*)d_in[2];
    const float* tail      = (const float*)d_in[3];
    const int*   pos_idx   = (const int*)d_in[4];
    const int*   neg_idx   = (const int*)d_in[5];
    const float* E         = (const float*)d_in[6];
    const float* wordE     = (const float*)d_in[7];
    const float* Wih_f     = (const float*)d_in[8];
    const float* Whh_f     = (const float*)d_in[9];
    const float* bih_f     = (const float*)d_in[10];
    const float* bhh_f     = (const float*)d_in[11];
    const float* Wih_b     = (const float*)d_in[12];
    const float* Whh_b     = (const float*)d_in[13];
    const float* bih_b     = (const float*)d_in[14];
    const float* bhh_b     = (const float*)d_in[15];
    const float* Wa        = (const float*)d_in[16];
    const float* va        = (const float*)d_in[17];
    const float* W_fc1     = (const float*)d_in[18];
    const float* b_fc1     = (const float*)d_in[19];
    const float* W_fc2     = (const float*)d_in[20];
    const float* head_bn_w = (const float*)d_in[21];
    const float* head_bn_b = (const float*)d_in[22];
    const float* score_bn_w= (const float*)d_in[23];
    const float* score_bn_b= (const float*)d_in[24];
    float* ws  = (float*)d_ws;
    short* sws = (short*)(ws + FLOAT_TOTAL);
    float* out = (float*)d_out;

    k_zero<<<(ZERO_N + 255) / 256, 256, 0, stream>>>(ws, ZERO_N);

    // one-time (per launch) bf16 weight conversion (+ Wa transposed to NT layout)
    k_f2bf<<<1024, 256, 0, stream>>>(Whh_f, sws + SOFF_WHH_F, 262144);
    k_f2bf<<<1024, 256, 0, stream>>>(Wih_f, sws + SOFF_WIH_F, 262144);
    k_f2bf<<<1024, 256, 0, stream>>>(Whh_b, sws + SOFF_WHH_B, 262144);
    k_f2bf<<<1024, 256, 0, stream>>>(Wih_b, sws + SOFF_WIH_B, 262144);
    k_f2bf_t<<<2048, 256, 0, stream>>>(Wa, sws + SOFF_WA, 512, 1024);
    k_f2bf<<<1024, 256, 0, stream>>>(W_fc1, sws + SOFF_WFC1,  262144);
    k_f2bf<<<800,  256, 0, stream>>>(W_fc2, sws + SOFF_WFC2,  204800);

    for (int t = 0; t < T_; ++t)
        k_lstm_step<<<dim3(16, 8, 2), 32, 0, stream>>>(
            wordE, question, qlen, bih_f, bhh_f, bih_b, bhh_b, ws, t);

    k_unrev<<<B_ * T_, 256, 0, stream>>>(qlen, ws);

    k_attn<<<dim3(64, 256), 32, 0, stream>>>(va, ws);
    k_softmax_ctx<<<B_, 256, 0, stream>>>(qlen, ws);

    // hidden = relu(ctx @ W_fc1^T + b_fc1) -> bf16 ; rel = hidden @ W_fc2^T -> fp32
    k_gemm_nt<<<dim3(32, 8), 32, 0, stream>>>(sws + SOFF_CTXB, 512, sws + SOFF_WFC1,
                                              512, b_fc1, nullptr, sws + SOFF_HIDB,
                                              512, 512, 1);
    k_gemm_nt<<<dim3(25, 8), 32, 0, stream>>>(sws + SOFF_HIDB, 512, sws + SOFF_WFC2,
                                              512, nullptr, ws + OFF_REL, nullptr,
                                              400, 512, 0);

    k_stats<<<1, 256, 0, stream>>>(E, head, ws, 0);
    k_res_ims<<<100, 256, 0, stream>>>(E, head, head_bn_w, head_bn_b, ws);
    k_stats<<<1, 256, 0, stream>>>(E, head, ws, 1);
    k_make_sn<<<(B_ * 416 + 255) / 256, 256, 0, stream>>>(score_bn_w, score_bn_b, ws);

    k_score_bce<<<NENT / 16, 256, 0, stream>>>(E, tail, ws);

    k_argmax_ans<<<B_, 256, 0, stream>>>(tail, E, ws);
    k_contrastive<<<B_, 256, 0, stream>>>(E, pos_idx, neg_idx, ws);
    k_finalize<<<1, 32, 0, stream>>>(ws, out);
}
